// EnhancedMessageLayer_40037685133359
// MI455X (gfx1250) — matrix operand order repaired and checked by reading, not compiled, not run on hardware
//
#include <hip/hip_runtime.h>
#include <hip/hip_bf16.h>

#define N_NODES 50000
#define N_EDGES 640000
#define H 128
#define KPAD1 288           // 2H + 3 padded to multiple of 32
#define ROWB1 (KPAD1 * 2)   // 576 bytes per msg_in row (f16)
#define ROWB_H (H * 2)      // 256 bytes per h row (f16)
#define ROWB_U (256 * 2)    // 512 bytes per upd_in row (f16)
#define OUTSTR 132          // padded f32 row stride (floats) for LN tile

typedef __attribute__((ext_vector_type(16))) _Float16 v16h;
typedef __attribute__((ext_vector_type(8)))  float    v8f;

union FragU { uint4 q[2]; v16h h; };

#define WAVE_LDS_FENCE() asm volatile("s_wait_dscnt 0" ::: "memory")

// ---------------------------------------------------------------------------
// Weight pre-swizzle: fp32 row-major W[Kact][128] -> f16 B-fragment layout.
// Fragment (c = K/32 chunk, t = N/16 tile): 32 lanes x 16 halves, where
// lane holds column n = 16t + (lane&15), halves q=0..15 map to
// k = 32c + 16*(lane>>4) + q  (contiguous k order, same as load_frag_a).
// ---------------------------------------------------------------------------
__global__ void swz_weights(const float* __restrict__ W, _Float16* __restrict__ out,
                            int Kact, int Kpad) {
    int idx = blockIdx.x * 256 + threadIdx.x;
    int total = Kpad * H;
    if (idx >= total) return;
    int q    = idx & 15;
    int lane = (idx >> 4) & 31;
    int t    = (idx >> 9) & 7;
    int c    = idx >> 12;
    int k = 32 * c + ((lane >> 4) << 4) + q;
    int n = t * 16 + (lane & 15);
    float v = (k < Kact) ? W[(size_t)k * H + n] : 0.0f;
    out[idx] = (_Float16)v;
}

__device__ __forceinline__ v16h load_frag_b(const _Float16* swz, int c, int t, int lane) {
    FragU f;
    const uint4* p = (const uint4*)(swz + ((size_t)(c * 8 + t) * 32 + lane) * 16);
    f.q[0] = p[0];
    f.q[1] = p[1];
    return f.h;
}

__device__ __forceinline__ v16h load_frag_a(const char* tile, int rowBytes, int c,
                                            int lm, int hi) {
    FragU f;
    const char* ap = tile + lm * rowBytes + c * 64 + hi * 32;
    f.q[0] = *(const uint4*)ap;
    f.q[1] = *(const uint4*)(ap + 16);
    return f.h;
}

__device__ __forceinline__ uint2 pack4h(float4 v) {
    union { _Float16 h[4]; uint2 u; } p;
    p.h[0] = (_Float16)v.x; p.h[1] = (_Float16)v.y;
    p.h[2] = (_Float16)v.z; p.h[3] = (_Float16)v.w;
    return p.u;
}

// ---------------------------------------------------------------------------
// Edge kernel: per wave one 32-edge tile (two 16-row A tiles sharing each
// B fragment -> halves weight traffic, 2 WMMAs per B load).
//   msg_in = [x[src] | x[dst] | attr | pad]  (32 x 288 f16 in LDS)
//   h   = relu(msg_in @ W1 + b1)             (32 x 128 f16 in LDS)
//   msg = h @ W2 + b2                        -> atomic add into agg[dst]
// ---------------------------------------------------------------------------
#define E_WAVE_LDS (32 * ROWB1 + 32 * ROWB_H)   // 18432 + 8192 = 26624
__global__ __launch_bounds__(64) void edge_kernel(
    const float* __restrict__ x, const int* __restrict__ ei,
    const float* __restrict__ eattr,
    const _Float16* __restrict__ w1s, const float* __restrict__ b1,
    const _Float16* __restrict__ w2s, const float* __restrict__ b2,
    float* __restrict__ agg) {
    __shared__ char smem[2 * E_WAVE_LDS];
    const int wave = threadIdx.x >> 5;
    const int lane = threadIdx.x & 31;
    char* msgin = smem + wave * E_WAVE_LDS;
    char* htile = msgin + 32 * ROWB1;

    int tile = blockIdx.x * 2 + wave;
    if (tile * 32 >= N_EDGES) return;
    const int e0 = tile * 32;
    const int* srcIdx = ei;
    const int* dstIdx = ei + N_EDGES;
    const int lm = lane & 15;
    const int hi = lane >> 4;

    // ---- stage msg_in tile (each lane supplies cols 4*lane..4*lane+3) ----
    for (int e = 0; e < 32; ++e) {
        int s = srcIdx[e0 + e];
        int d = dstIdx[e0 + e];
        float4 xs = *(const float4*)(x + (size_t)s * H + lane * 4);
        float4 xd = *(const float4*)(x + (size_t)d * H + lane * 4);
        *(uint2*)(msgin + e * ROWB1 + lane * 8)       = pack4h(xs);
        *(uint2*)(msgin + e * ROWB1 + 256 + lane * 8) = pack4h(xd);
    }
    {   // edge_attr (3) + zero pad up to col 288; one row per lane
        const float* ap = eattr + (size_t)(e0 + lane) * 3;
        union { _Float16 h[8]; uint4 u; } p;
        p.h[0] = (_Float16)ap[0]; p.h[1] = (_Float16)ap[1]; p.h[2] = (_Float16)ap[2];
        p.h[3] = (_Float16)0.f; p.h[4] = (_Float16)0.f; p.h[5] = (_Float16)0.f;
        p.h[6] = (_Float16)0.f; p.h[7] = (_Float16)0.f;
        char* rp = msgin + lane * ROWB1 + 512;
        uint4 z = make_uint4(0u, 0u, 0u, 0u);
        *(uint4*)(rp)      = p.u;
        *(uint4*)(rp + 16) = z;
        *(uint4*)(rp + 32) = z;
        *(uint4*)(rp + 48) = z;
    }
    // destination nodes per accumulator row (M = r + 8*hi) for both sub-tiles
    int dn0[8], dn1[8];
    for (int r = 0; r < 8; ++r) {
        dn0[r] = dstIdx[e0 + hi * 8 + r];
        dn1[r] = dstIdx[e0 + 16 + hi * 8 + r];
    }
    WAVE_LDS_FENCE();

    // ---- GEMM1: 9 K-chunks, 8 N-tiles, 2 M-tiles; relu -> h tile ----
    for (int t = 0; t < 8; ++t) {
        v8f acc0 = {}, acc1 = {};
        for (int c = 0; c < 9; ++c) {
            v16h b  = load_frag_b(w1s, c, t, lane);
            v16h a0 = load_frag_a(msgin, ROWB1, c, lm, hi);
            v16h a1 = load_frag_a(msgin + 16 * ROWB1, ROWB1, c, lm, hi);
            acc0 = __builtin_amdgcn_wmma_f32_16x16x32_f16(false, a0, false, b,
                                                          (short)0, acc0, false, false);
            acc1 = __builtin_amdgcn_wmma_f32_16x16x32_f16(false, a1, false, b,
                                                          (short)0, acc1, false, false);
        }
        float bias = b1[t * 16 + lm];
        for (int r = 0; r < 8; ++r) {
            float v0 = acc0[r] + bias; v0 = v0 > 0.f ? v0 : 0.f;
            float v1 = acc1[r] + bias; v1 = v1 > 0.f ? v1 : 0.f;
            int row = r + hi * 8;
            *(_Float16*)(htile + row * ROWB_H + (t * 16 + lm) * 2)        = (_Float16)v0;
            *(_Float16*)(htile + (row + 16) * ROWB_H + (t * 16 + lm) * 2) = (_Float16)v1;
        }
    }
    WAVE_LDS_FENCE();

    // ---- GEMM2: 4 K-chunks, 8 N-tiles, 2 M-tiles; scatter-add into agg ----
    for (int t = 0; t < 8; ++t) {
        v8f acc0 = {}, acc1 = {};
        for (int c = 0; c < 4; ++c) {
            v16h b  = load_frag_b(w2s, c, t, lane);
            v16h a0 = load_frag_a(htile, ROWB_H, c, lm, hi);
            v16h a1 = load_frag_a(htile + 16 * ROWB_H, ROWB_H, c, lm, hi);
            acc0 = __builtin_amdgcn_wmma_f32_16x16x32_f16(false, a0, false, b,
                                                          (short)0, acc0, false, false);
            acc1 = __builtin_amdgcn_wmma_f32_16x16x32_f16(false, a1, false, b,
                                                          (short)0, acc1, false, false);
        }
        float bias = b2[t * 16 + lm];
        for (int r = 0; r < 8; ++r) {
            unsafeAtomicAdd(agg + (size_t)dn0[r] * H + t * 16 + lm, acc0[r] + bias);
            unsafeAtomicAdd(agg + (size_t)dn1[r] * H + t * 16 + lm, acc1[r] + bias);
        }
    }
}

// ---------------------------------------------------------------------------
// Node kernel: per wave one 16-node tile.
//   upd_in = [x | agg] (16 x 256 f16); gate = sigmoid(upd_in@Wg+bg)
//   update = relu(upd_in@Wu1+bu1)@Wu2+bu2; out = g*u + (1-g)*x; LayerNorm.
// x (f32) is staged into the out tile and overwritten in place.
// ---------------------------------------------------------------------------
#define N_WAVE_LDS (16 * ROWB_U + 16 * ROWB_H + 16 * OUTSTR * 4 + 256)
__global__ __launch_bounds__(64) void node_kernel(
    const float* __restrict__ x, const float* __restrict__ agg,
    const _Float16* __restrict__ wgs, const float* __restrict__ bg,
    const _Float16* __restrict__ wu1s, const float* __restrict__ bu1,
    const _Float16* __restrict__ wu2s, const float* __restrict__ bu2,
    const float* __restrict__ gamma, const float* __restrict__ beta,
    float* __restrict__ dout) {
    __shared__ char smem[2 * N_WAVE_LDS];
    const int wave = threadIdx.x >> 5;
    const int lane = threadIdx.x & 31;
    char* updin = smem + wave * N_WAVE_LDS;
    char* htile = updin + 16 * ROWB_U;
    float* outf = (float*)(htile + 16 * ROWB_H);
    float* lnb  = outf + 16 * OUTSTR;

    int tile = blockIdx.x * 2 + wave;
    if (tile * 16 >= N_NODES) return;
    const int n0 = tile * 16;
    const int lm = lane & 15;
    const int hi = lane >> 4;

    // ---- stage [x | agg] as f16; x also as f32 into out tile ----
    for (int n = 0; n < 16; ++n) {
        float4 xv = *(const float4*)(x   + (size_t)(n0 + n) * H + lane * 4);
        float4 av = *(const float4*)(agg + (size_t)(n0 + n) * H + lane * 4);
        *(uint2*)(updin + n * ROWB_U + lane * 8)       = pack4h(xv);
        *(uint2*)(updin + n * ROWB_U + 256 + lane * 8) = pack4h(av);
        *(float4*)(outf + n * OUTSTR + lane * 4) = xv;
    }
    WAVE_LDS_FENCE();

    // ---- phase 1: h = relu(upd_in @ Wu1 + bu1) ----
    for (int t = 0; t < 8; ++t) {
        v8f acc = {};
        for (int c = 0; c < 8; ++c) {
            v16h a = load_frag_a(updin, ROWB_U, c, lm, hi);
            v16h b = load_frag_b(wu1s, c, t, lane);
            acc = __builtin_amdgcn_wmma_f32_16x16x32_f16(false, a, false, b,
                                                         (short)0, acc, false, false);
        }
        float bias = bu1[t * 16 + lm];
        for (int r = 0; r < 8; ++r) {
            float v = acc[r] + bias;
            v = v > 0.f ? v : 0.f;
            *(_Float16*)(htile + (r + hi * 8) * ROWB_H + (t * 16 + lm) * 2) = (_Float16)v;
        }
    }
    WAVE_LDS_FENCE();

    // ---- phase 2: gate + update, blend with x (in-place in out tile) ----
    for (int t = 0; t < 8; ++t) {
        v8f accu = {};
        for (int c = 0; c < 4; ++c) {
            v16h a = load_frag_a(htile, ROWB_H, c, lm, hi);
            v16h b = load_frag_b(wu2s, c, t, lane);
            accu = __builtin_amdgcn_wmma_f32_16x16x32_f16(false, a, false, b,
                                                          (short)0, accu, false, false);
        }
        v8f accg = {};
        for (int c = 0; c < 8; ++c) {
            v16h a = load_frag_a(updin, ROWB_U, c, lm, hi);
            v16h b = load_frag_b(wgs, c, t, lane);
            accg = __builtin_amdgcn_wmma_f32_16x16x32_f16(false, a, false, b,
                                                          (short)0, accg, false, false);
        }
        float biasg = bg[t * 16 + lm];
        float biasu = bu2[t * 16 + lm];
        for (int r = 0; r < 8; ++r) {
            int row = r + hi * 8;
            float g = 1.0f / (1.0f + __expf(-(accg[r] + biasg)));
            float u = accu[r] + biasu;
            float xv = outf[row * OUTSTR + t * 16 + lm];
            outf[row * OUTSTR + t * 16 + lm] = g * u + (1.0f - g) * xv;
        }
    }
    WAVE_LDS_FENCE();

    // ---- phase 3: LayerNorm. Two lanes per row (64 cols each). ----
    {
        int row = lm, half = hi;
        const float* rp = outf + row * OUTSTR + half * 64;
        float s1 = 0.f, s2 = 0.f;
        for (int i = 0; i < 64; ++i) { float v = rp[i]; s1 += v; s2 += v * v; }
        lnb[lane * 2] = s1; lnb[lane * 2 + 1] = s2;
        WAVE_LDS_FENCE();
        s1 += lnb[(lane ^ 16) * 2];
        s2 += lnb[(lane ^ 16) * 2 + 1];
        float mean = s1 * (1.0f / 128.0f);
        float var  = s2 * (1.0f / 128.0f) - mean * mean;
        float inv  = rsqrtf(var + 1e-5f);
        WAVE_LDS_FENCE();            // all partner reads done before overwrite
        if (lane < 16) { lnb[row * 2] = mean; lnb[row * 2 + 1] = inv; }
        WAVE_LDS_FENCE();
    }
    float4 g4 = *(const float4*)(gamma + lane * 4);
    float4 b4 = *(const float4*)(beta  + lane * 4);
    for (int n = 0; n < 16; ++n) {
        float mean = lnb[n * 2], inv = lnb[n * 2 + 1];
        float4 o = *(const float4*)(outf + n * OUTSTR + lane * 4);
        float4 r;
        r.x = (o.x - mean) * inv * g4.x + b4.x;
        r.y = (o.y - mean) * inv * g4.y + b4.y;
        r.z = (o.z - mean) * inv * g4.z + b4.z;
        r.w = (o.w - mean) * inv * g4.w + b4.w;
        *(float4*)(dout + (size_t)(n0 + n) * H + lane * 4) = r;
    }
}

// ---------------------------------------------------------------------------
extern "C" void kernel_launch(void* const* d_in, const int* in_sizes, int n_in,
                              void* d_out, int out_size, void* d_ws, size_t ws_size,
                              hipStream_t stream) {
    const float* x     = (const float*)d_in[0];
    const int*   ei    = (const int*)d_in[1];
    const float* eattr = (const float*)d_in[2];
    const float* Wm1 = (const float*)d_in[3];
    const float* bm1 = (const float*)d_in[4];
    const float* Wm2 = (const float*)d_in[5];
    const float* bm2 = (const float*)d_in[6];
    const float* Wg  = (const float*)d_in[7];
    const float* bg  = (const float*)d_in[8];
    const float* Wu1 = (const float*)d_in[9];
    const float* bu1 = (const float*)d_in[10];
    const float* Wu2 = (const float*)d_in[11];
    const float* bu2 = (const float*)d_in[12];
    const float* gam = (const float*)d_in[13];
    const float* bet = (const float*)d_in[14];
    float* out = (float*)d_out;

    char* ws = (char*)d_ws;
    float* agg = (float*)ws;
    size_t off = (size_t)N_NODES * H * sizeof(float);      // 25.6 MB
    _Float16* w1s  = (_Float16*)(ws + off); off += (size_t)KPAD1 * H * 2;
    _Float16* w2s  = (_Float16*)(ws + off); off += (size_t)H * H * 2;
    _Float16* wgs  = (_Float16*)(ws + off); off += (size_t)256 * H * 2;
    _Float16* wu1s = (_Float16*)(ws + off); off += (size_t)256 * H * 2;
    _Float16* wu2s = (_Float16*)(ws + off); off += (size_t)H * H * 2;

    hipMemsetAsync(agg, 0, (size_t)N_NODES * H * sizeof(float), stream);
    swz_weights<<<(KPAD1 * H + 255) / 256, 256, 0, stream>>>(Wm1, w1s, 2 * H + 3, KPAD1);
    swz_weights<<<(H * H + 255) / 256,     256, 0, stream>>>(Wm2, w2s, H, H);
    swz_weights<<<(256 * H + 255) / 256,   256, 0, stream>>>(Wg,  wgs, 256, 256);
    swz_weights<<<(256 * H + 255) / 256,   256, 0, stream>>>(Wu1, wu1s, 256, 256);
    swz_weights<<<(H * H + 255) / 256,     256, 0, stream>>>(Wu2, wu2s, H, H);

    edge_kernel<<<(N_EDGES / 32 + 1) / 2, 64, 0, stream>>>(x, ei, eattr,
                                                           w1s, bm1, w2s, bm2, agg);
    node_kernel<<<(N_NODES / 16 + 1) / 2, 64, 0, stream>>>(x, agg, wgs, bg,
                                                           wu1s, bu1, wu2s, bu2,
                                                           gam, bet, out);
}
